// AttnLayer_3332894621898
// MI455X (gfx1250) — compile-verified, matrix operand order checked by reading, not run on hardware
//
#include <hip/hip_runtime.h>
#include <hip/hip_bf16.h>
#include <cstdint>

// ---------------------------------------------------------------------------
// GATv2 layer for gfx1250 (MI455X).
//   N=50000 nodes, E=1.6M edges, IN=128, HEADS=4, C=32 (HC=128)
// Pipeline:
//   0) init: out=bias, amax=-inf, asum=0, loop_attr=0
//   1) segment-max edge_weight by dst -> loop_attr
//   2) WMMA GEMM: x_l = x@Wl+bl, x_r = x@Wr+br   (v_wmma_f32_16x16x4_f32)
//   3) per-edge alpha: wave32 per edge, per-head shfl reduction, atomic-max
//   4) exp + atomic-add segment sums
//   5) aggregate: out[dst] += alpha * x_l[src]    (global_atomic_add_f32)
// ---------------------------------------------------------------------------

#define NEG_SLOPE 0.2f

typedef float v2f __attribute__((ext_vector_type(2)));
typedef float v8f __attribute__((ext_vector_type(8)));
typedef _Float16 v16h __attribute__((ext_vector_type(16)));

// float atomic max that works for positive and negative values
// (amax initialized to -inf). Non-negative floats order like ints; negative
// floats order reversed as uints.
__device__ __forceinline__ void atomicMaxFloat(float* addr, float value) {
  if (value >= 0.0f) {
    atomicMax((int*)addr, __float_as_int(value));
  } else {
    atomicMin((unsigned int*)addr, __float_as_uint(value));
  }
}

// --------------------------------------------------------------------------
// 0) init: out = bias (atomics accumulate on top), amax=-inf, asum=0, loop=0
// --------------------------------------------------------------------------
__global__ void init_kernel(float* __restrict__ out, const float* __restrict__ bias,
                            float* __restrict__ amax, float* __restrict__ asum,
                            float* __restrict__ loop_attr, int n) {
  int idx = blockIdx.x * blockDim.x + threadIdx.x;
  int tot = n * 128;
  if (idx < tot) out[idx] = bias[idx & 127];
  if (idx < n * 4) {
    amax[idx] = -__builtin_inff();
    asum[idx] = 0.0f;
  }
  if (idx < n) loop_attr[idx] = 0.0f;
}

// --------------------------------------------------------------------------
// 1) loop_attr[d] = max(0, max_{e: dst[e]==d} w[e])   (weights are >= 0)
// --------------------------------------------------------------------------
__global__ void segmax_kernel(const int* __restrict__ ei, const float* __restrict__ ew,
                              float* __restrict__ loop_attr, int E) {
  int t = blockIdx.x * blockDim.x + threadIdx.x;
  if (t >= E) return;
  int d = ei[E + t];
  atomicMaxFloat(&loop_attr[d], ew[t]);
}

// --------------------------------------------------------------------------
// 2) GEMM via WMMA. One wave -> one 16x16 output tile, K=128.
//    blockDim=256 (8 waves): wave w handles column tile w (8*16 = 128 cols).
//    grid = (ceil(n/16), 2): y==0 -> Wl/bl -> xl ; y==1 -> Wr/br -> xr
// --------------------------------------------------------------------------
__global__ __launch_bounds__(256) void gemm_wmma_kernel(
    const float* __restrict__ x,
    const float* __restrict__ Wl, const float* __restrict__ bl,
    const float* __restrict__ Wr, const float* __restrict__ br,
    float* __restrict__ xl, float* __restrict__ xr, int n) {
  const int wave = threadIdx.x >> 5;
  const int lane = threadIdx.x & 31;
  const int row0 = blockIdx.x * 16;
  const float* W;
  const float* b;
  float* out;
  if (blockIdx.y == 0) { W = Wl; b = bl; out = xl; }
  else                 { W = Wr; b = br; out = xr; }
  const int col0 = wave * 16;
  const int m = lane & 15;   // row (A) / col (B,C,D) index within tile
  const int q = lane >> 4;   // half-select

  // clamp row for safety (n is a multiple of 16 here, stores are guarded)
  int rA = row0 + m;
  if (rA >= n) rA = n - 1;
  const float* xrow = x + (size_t)rA * 128;

  v8f acc = {};

#if __has_builtin(__builtin_amdgcn_wmma_f32_16x16x4_f32)
  // Exact fp32 path: A is 16x4 (2 floats/lane), B is 4x16 (2 floats/lane).
  #pragma unroll
  for (int kk = 0; kk < 128; kk += 4) {
    const int k = kk + q * 2;
    v2f a;
    a.x = xrow[k];
    a.y = xrow[k + 1];
    v2f bm;
    bm.x = W[(size_t)k * 128 + col0 + m];
    bm.y = W[(size_t)(k + 1) * 128 + col0 + m];
    acc = __builtin_amdgcn_wmma_f32_16x16x4_f32(false, a, false, bm,
                                                (short)0, acc, false, false);
  }
#else
  // Fallback: f16 inputs, f32 accumulate (v_wmma_f32_16x16x32_f16).
  #pragma unroll
  for (int kk = 0; kk < 128; kk += 32) {
    v16h a, bm;
    #pragma unroll
    for (int i = 0; i < 16; ++i) {
      const int k = kk + ((i < 8) ? (q * 8 + i) : (16 + q * 8 + (i - 8)));
      a[i]  = (_Float16)xrow[k];
      bm[i] = (_Float16)W[(size_t)k * 128 + col0 + m];
    }
    acc = __builtin_amdgcn_wmma_f32_16x16x32_f16(false, a, false, bm,
                                                 (short)0, acc, false, false);
  }
#endif

  const float bv = b[col0 + m];
  #pragma unroll
  for (int v = 0; v < 8; ++v) {
    const int rr = row0 + v + 8 * q;
    if (rr < n) out[(size_t)rr * 128 + col0 + m] = acc[v] + bv;
  }
}

// --------------------------------------------------------------------------
// 3) alpha pass: one wave32 per edge. lane handles hc = j*32+lane, j = head.
//    alpha[e][h] = sum_c leaky(xl[s][hc]+xr[d][hc]+w*We[hc]) * att[hc]
//    also atomic-max into amax[d][h].
// --------------------------------------------------------------------------
__global__ __launch_bounds__(256) void edge_alpha_kernel(
    const int* __restrict__ ei, const float* __restrict__ ew,
    const float* __restrict__ loop_attr,
    const float* __restrict__ xl, const float* __restrict__ xr,
    const float* __restrict__ We, const float* __restrict__ att,
    float* __restrict__ alpha, float* __restrict__ amax, int E, int n) {
  const int e = blockIdx.x * 8 + (threadIdx.x >> 5);
  const int EN = E + n;
  if (e >= EN) return;
  const int lane = threadIdx.x & 31;

  int s, d;
  float w;
  if (e < E) { s = ei[e]; d = ei[E + e]; w = ew[e]; }
  else       { s = e - E; d = s;         w = loop_attr[s]; }

  const float* pl = xl + (size_t)s * 128;
  const float* pr = xr + (size_t)d * 128;

  float sums[4];
  #pragma unroll
  for (int j = 0; j < 4; ++j) {
    const int hc = j * 32 + lane;
    float v = pl[hc] + pr[hc] + w * We[hc];
    v = (v > 0.0f) ? v : NEG_SLOPE * v;
    float val = v * att[hc];
    #pragma unroll
    for (int off = 16; off > 0; off >>= 1) val += __shfl_xor(val, off, 32);
    sums[j] = val;
  }

  if (lane == 0) {
    #pragma unroll
    for (int j = 0; j < 4; ++j) {
      alpha[(size_t)e * 4 + j] = sums[j];
      atomicMaxFloat(&amax[(size_t)d * 4 + j], sums[j]);
    }
  }
}

// --------------------------------------------------------------------------
// 4) aexp = exp(alpha - amax[dst]); asum[dst] += aexp  (one thread per (e,h))
// --------------------------------------------------------------------------
__global__ void edge_exp_kernel(const int* __restrict__ ei,
                                const float* __restrict__ amax,
                                float* __restrict__ alpha,
                                float* __restrict__ asum, int E, int n) {
  const long long t = (long long)blockIdx.x * blockDim.x + threadIdx.x;
  const long long tot = (long long)(E + n) * 4;
  if (t >= tot) return;
  const int e = (int)(t >> 2);
  const int h = (int)(t & 3);
  const int d = (e < E) ? ei[E + e] : (e - E);
  const float ae = expf(alpha[t] - amax[(size_t)d * 4 + h]);
  alpha[t] = ae;
  unsafeAtomicAdd(&asum[(size_t)d * 4 + h], ae);
}

// --------------------------------------------------------------------------
// 5) out[d][hc] += (aexp[e][h] / (asum[d][h]+1e-16)) * xl[s][hc]
//    one wave32 per edge; 4 native f32 atomics per lane.
// --------------------------------------------------------------------------
__global__ __launch_bounds__(256) void edge_aggr_kernel(
    const int* __restrict__ ei, const float* __restrict__ xl,
    const float* __restrict__ alpha, const float* __restrict__ asum,
    float* __restrict__ out, int E, int n) {
  const int e = blockIdx.x * 8 + (threadIdx.x >> 5);
  if (e >= E + n) return;
  const int lane = threadIdx.x & 31;

  int s, d;
  if (e < E) { s = ei[e]; d = ei[E + e]; }
  else       { s = e - E; d = s; }

  const float4 a4 = ((const float4*)alpha)[e];
  const float4 s4 = ((const float4*)asum)[d];
  const float c0 = a4.x / (s4.x + 1e-16f);
  const float c1 = a4.y / (s4.y + 1e-16f);
  const float c2 = a4.z / (s4.z + 1e-16f);
  const float c3 = a4.w / (s4.w + 1e-16f);

  const float* pl = xl + (size_t)s * 128;
  float* po = out + (size_t)d * 128;
  unsafeAtomicAdd(&po[lane],      pl[lane]      * c0);
  unsafeAtomicAdd(&po[32 + lane], pl[32 + lane] * c1);
  unsafeAtomicAdd(&po[64 + lane], pl[64 + lane] * c2);
  unsafeAtomicAdd(&po[96 + lane], pl[96 + lane] * c3);
}

// ---------------------------------------------------------------------------
// Host launcher
// ---------------------------------------------------------------------------
extern "C" void kernel_launch(void* const* d_in, const int* in_sizes, int n_in,
                              void* d_out, int out_size, void* d_ws, size_t ws_size,
                              hipStream_t stream) {
  const float* x    = (const float*)d_in[0];
  const int*   ei   = (const int*)d_in[1];   // [2,E] flat: src then dst
  const float* ew   = (const float*)d_in[2];
  const float* Wl   = (const float*)d_in[3];
  const float* bl   = (const float*)d_in[4];
  const float* Wr   = (const float*)d_in[5];
  const float* br   = (const float*)d_in[6];
  const float* We   = (const float*)d_in[7];  // [1,128]
  const float* att  = (const float*)d_in[8];  // [4,32]
  const float* bias = (const float*)d_in[9];  // [128]
  float* out = (float*)d_out;

  const int N = in_sizes[0] / 128;
  const int E = in_sizes[2];
  const int EN = E + N;

  // workspace layout (floats)
  float* ws = (float*)d_ws;
  float* xl        = ws;                         // N*128
  float* xr        = xl + (size_t)N * 128;       // N*128
  float* alpha     = xr + (size_t)N * 128;       // EN*4 (16B aligned: N*256 mult of 4)
  float* amax      = alpha + (size_t)EN * 4;     // N*4
  float* asum      = amax + (size_t)N * 4;       // N*4
  float* loop_attr = asum + (size_t)N * 4;       // N

  // 0) init
  {
    int tot = N * 128;
    init_kernel<<<(tot + 255) / 256, 256, 0, stream>>>(out, bias, amax, asum,
                                                       loop_attr, N);
  }
  // 1) segment max
  segmax_kernel<<<(E + 255) / 256, 256, 0, stream>>>(ei, ew, loop_attr, E);
  // 2) GEMMs (WMMA)
  {
    dim3 grid((N + 15) / 16, 2);
    gemm_wmma_kernel<<<grid, 256, 0, stream>>>(x, Wl, bl, Wr, br, xl, xr, N);
  }
  // 3) alpha
  edge_alpha_kernel<<<(EN + 7) / 8, 256, 0, stream>>>(ei, ew, loop_attr, xl, xr,
                                                      We, att, alpha, amax, E, N);
  // 4) exp + sums
  {
    long long tot = (long long)EN * 4;
    int blocks = (int)((tot + 255) / 256);
    edge_exp_kernel<<<blocks, 256, 0, stream>>>(ei, amax, alpha, asum, E, N);
  }
  // 5) aggregate
  edge_aggr_kernel<<<(EN + 7) / 8, 256, 0, stream>>>(ei, xl, alpha, asum, out, E, N);
}
